// InteractionNet_39539468927641
// MI455X (gfx1250) — matrix operand order repaired and checked by reading, not compiled, not run on hardware
//
#include <hip/hip_runtime.h>

#define N_NODES 20000
#define N_EDGES 640000
#define NF 64
#define NB 20

typedef __attribute__((ext_vector_type(16))) __bf16        v16bf;
typedef __attribute__((ext_vector_type(8)))  float         v8f;
typedef __attribute__((ext_vector_type(8)))  unsigned int  v8u;

union BF16x16 { v8u u; v16bf b; };

// fast SiLU: x * v_rcp_f32(1 + e^-x)  (avoids the IEEE div_scale/div_fixup chain)
__device__ __forceinline__ float silu_f(float x) {
    return x * __builtin_amdgcn_rcpf(1.0f + __expf(-x));
}

// round-to-nearest-even f32 -> bf16 (raw ushort)
__device__ __forceinline__ unsigned int f2bf_u(float f) {
    unsigned int u = __float_as_uint(f);
    u += 0x7FFFu + ((u >> 16) & 1u);
    return u >> 16;
}
__device__ __forceinline__ unsigned short f2bf(float f) {
    return (unsigned short)f2bf_u(f);
}

// ---------------------------------------------------------------------------
// Kernel 1: mn = act(atom@W1+b1)@W2+b2 ; also initialize d_out with copies of
// atom_node / force_node so edge_kernel atomics accumulate on top.
// ---------------------------------------------------------------------------
__global__ void __launch_bounds__(NF) node_mlp_kernel(
    const float* __restrict__ atom, const float* __restrict__ force,
    const float* __restrict__ W1, const float* __restrict__ b1,
    const float* __restrict__ W2, const float* __restrict__ b2,
    float* __restrict__ mn, float* __restrict__ atom_out, float* __restrict__ force_out)
{
    __shared__ float xs[NF];
    __shared__ float hs[NF];
    const int n = blockIdx.x, t = threadIdx.x;
    float x = atom[n * NF + t];
    xs[t] = x;
    atom_out[n * NF + t] = x;                       // init output accumulator
    #pragma unroll
    for (int d = 0; d < 3; ++d)                     // init force output accumulator
        force_out[n * 3 * NF + d * NF + t] = force[n * 3 * NF + d * NF + t];
    __syncthreads();

    float h = b1[t];
    #pragma unroll 8
    for (int k = 0; k < NF; ++k) h = fmaf(xs[k], W1[k * NF + t], h);
    h = silu_f(h);
    hs[t] = h;
    __syncthreads();

    float o = b2[t];
    #pragma unroll 8
    for (int k = 0; k < NF; ++k) o = fmaf(hs[k], W2[k * NF + t], o);
    mn[n * NF + t] = o;
}

// ---------------------------------------------------------------------------
// WMMA fragment loaders (bf16, 16x16x32). Tiles stored in LDS with row stride
// 66 halves (132B) so 16 consecutive rows map to distinct banks.
// A (16x32, 16-bit): lane m=L&15; vgpr v<4: K=2v,2v+1 (+8 for hi half-wave);
//                    vgpr v>=4: K=16+2(v-4),(+1) (+8 for hi half-wave).
// B (32x16, 16-bit): stored TRANSPOSED in LDS as [n][k]; lane n=L&15;
//                    same K mapping as A (v<4: K=2v,2v+1; v>=4: K=16+2(v-4),(+1); +8 for hi half-wave).
// ---------------------------------------------------------------------------
__device__ __forceinline__ v16bf load_afrag(const unsigned short* tile, int lane, int k0)
{
    BF16x16 r;
    const int m   = lane & 15;
    const int hi8 = (lane >> 4) << 3;
    const unsigned short* row = tile + m * 66 + k0;
    #pragma unroll
    for (int v = 0; v < 8; ++v) {
        const int kk = ((v < 4) ? (2 * v) : (16 + 2 * (v - 4))) + hi8;
        r.u[v] = *(const unsigned int*)(row + kk);
    }
    return r.b;
}

__device__ __forceinline__ v16bf load_bfrag(const unsigned short* wt, int lane, int nAbs, int k0)
{
    BF16x16 r;
    const int hi8 = (lane >> 4) << 3;
    const unsigned short* row = wt + nAbs * 66 + k0;
    #pragma unroll
    for (int v = 0; v < 8; ++v)
        r.u[v] = *(const unsigned int*)(row + ((v < 4) ? (2 * v) : (16 + 2 * (v - 4))) + hi8);
    return r.b;
}

// ---------------------------------------------------------------------------
// Kernel 2: edge pipeline. 128 threads = 4 waves; each wave processes 4 tiles
// of 16 edges. Per tile: VALU message (K=20 basis GEMM * gathered node parts),
// f32 atomic scatter of message, then 4 bf16 WMMA GEMMs (2-layer MLPs), SiLU,
// and the equivariant combine + f32 atomic scatter.
// ---------------------------------------------------------------------------
__global__ void __launch_bounds__(128) edge_kernel(
    const float* __restrict__ mn, const float* __restrict__ force_in,
    const float* __restrict__ dir_edge, const float* __restrict__ dist_edge,
    const long long* __restrict__ edge_index, const float* __restrict__ We,
    const float* __restrict__ Wq1a, const float* __restrict__ Wq1b,
    const float* __restrict__ Wq2a, const float* __restrict__ Wq2b,
    float* atom_out, float* force_out)
{
    __shared__ unsigned short Wt1a[NF * 66];
    __shared__ unsigned short Wt1b[NF * 66];
    __shared__ unsigned short Wt2a[NF * 66];
    __shared__ unsigned short Wt2b[NF * 66];
    __shared__ unsigned short msgT[4][16 * 66];
    __shared__ unsigned short h1T[4][16 * 66];
    __shared__ unsigned short h2T[4][16 * 66];
    __shared__ long long sidxS[4][16];
    __shared__ long long didxS[4][16];
    __shared__ float dirsS[4][16][3];

    const int tid = threadIdx.x;
    const int w = tid >> 5;
    const int lane = tid & 31;

    // Stage all four 64x64 weight matrices as bf16, transposed to [n][k].
    for (int idx = tid; idx < NF * NF; idx += blockDim.x) {
        const int k = idx >> 6, n = idx & 63;
        const int o = n * 66 + k;
        Wt1a[o] = f2bf(Wq1a[idx]);
        Wt1b[o] = f2bf(Wq1b[idx]);
        Wt2a[o] = f2bf(Wq2a[idx]);
        Wt2b[o] = f2bf(Wq2b[idx]);
    }
    __syncthreads();

    unsigned short* msg = msgT[w];
    unsigned short* h1  = h1T[w];
    unsigned short* h2  = h2T[w];

    for (int t = 0; t < 4; ++t) {
        const int tile = blockIdx.x * 16 + w * 4 + t;
        const int e0 = tile * 16;

        __syncthreads();   // protect LDS tiles across iterations (uniform trip count)
        if (lane < 16) {
            const int e = e0 + lane;
            sidxS[w][lane] = edge_index[e];
            didxS[w][lane] = edge_index[N_EDGES + e];
            dirsS[w][lane][0] = dir_edge[e * 3 + 0];
            dirsS[w][lane][1] = dir_edge[e * 3 + 1];
            dirsS[w][lane][2] = dir_edge[e * 3 + 2];
        }
        __syncthreads();

        // ---- Phase 1: message = (dist@We) * mn[src] * mn[dst]  (f32 VALU) ----
        {
            const int r = lane & 15;
            const int fbase = (lane >> 4) * 32;
            const int e = e0 + r;
            const int s  = (int)sidxS[w][r];
            const int d2 = (int)didxS[w][r];
            float dd[NB];
            #pragma unroll
            for (int b = 0; b < NB; ++b) dd[b] = dist_edge[(long)e * NB + b];
            const float* mns = mn + s * NF;
            const float* mnd = mn + d2 * NF;
            float* aout = atom_out + s * NF;
            #pragma unroll 4
            for (int j = 0; j < 32; j += 2) {       // two features at a time -> packed bf16 store
                const int f = fbase + j;
                float me0 = 0.f, me1 = 0.f;
                #pragma unroll
                for (int b = 0; b < NB; ++b) {
                    const float db = dd[b];
                    me0 = fmaf(db, We[b * NF + f], me0);
                    me1 = fmaf(db, We[b * NF + f + 1], me1);
                }
                const float v0 = me0 * mns[f] * mnd[f];
                const float v1 = me1 * mns[f + 1] * mnd[f + 1];
                *(unsigned int*)(msg + r * 66 + f) = f2bf_u(v0) | (f2bf_u(v1) << 16);
                atomicAdd(aout + f, v0);            // inv_update1 scatter (L2-resident)
                atomicAdd(aout + f + 1, v1);
            }
        }
        __syncthreads();

        // ---- Phase 2: h1 = silu(msg@Wq1a), h2 = silu(msg@Wq2a)  (WMMA) ----
        #pragma unroll
        for (int nt = 0; nt < 4; ++nt) {
            v8f a1 = {}; v8f a2 = {};
            const int nAbs = nt * 16 + (lane & 15);
            #pragma unroll
            for (int kc = 0; kc < 2; ++kc) {
                const v16bf af  = load_afrag(msg, lane, kc * 32);
                const v16bf bf1 = load_bfrag(Wt1a, lane, nAbs, kc * 32);
                const v16bf bf2 = load_bfrag(Wt2a, lane, nAbs, kc * 32);
                a1 = __builtin_amdgcn_wmma_f32_16x16x32_bf16(false, af, false, bf1, (short)0, a1, false, false);
                a2 = __builtin_amdgcn_wmma_f32_16x16x32_bf16(false, af, false, bf2, (short)0, a2, false, false);
            }
            const int hi8 = (lane >> 4) << 3;
            #pragma unroll
            for (int i = 0; i < 8; ++i) {
                h1[(i + hi8) * 66 + nAbs] = f2bf(silu_f(a1[i]));
                h2[(i + hi8) * 66 + nAbs] = f2bf(silu_f(a2[i]));
            }
        }
        __syncthreads();

        // ---- Phase 3: em1 = h1@Wq1b, em2 = h2@Wq2b (WMMA), then
        //      scatter em1*dir + em2*force_in[dst] into force_out[src] ----
        #pragma unroll
        for (int nt = 0; nt < 4; ++nt) {
            v8f e1v = {}; v8f e2v = {};
            const int nAbs = nt * 16 + (lane & 15);
            #pragma unroll
            for (int kc = 0; kc < 2; ++kc) {
                const v16bf a1f = load_afrag(h1, lane, kc * 32);
                const v16bf a2f = load_afrag(h2, lane, kc * 32);
                const v16bf b1f = load_bfrag(Wt1b, lane, nAbs, kc * 32);
                const v16bf b2f = load_bfrag(Wt2b, lane, nAbs, kc * 32);
                e1v = __builtin_amdgcn_wmma_f32_16x16x32_bf16(false, a1f, false, b1f, (short)0, e1v, false, false);
                e2v = __builtin_amdgcn_wmma_f32_16x16x32_bf16(false, a2f, false, b2f, (short)0, e2v, false, false);
            }
            const int hi8 = (lane >> 4) << 3;
            #pragma unroll
            for (int i = 0; i < 8; ++i) {
                const int m = i + hi8;
                const int s  = (int)sidxS[w][m];
                const int d2 = (int)didxS[w][m];
                const float e1 = e1v[i], e2 = e2v[i];
                const float* fin = force_in + d2 * 3 * NF + nAbs;   // original force_node[dst]
                float* fout = force_out + s * 3 * NF + nAbs;
                #pragma unroll
                for (int d = 0; d < 3; ++d) {
                    const float val = fmaf(e1, dirsS[w][m][d], e2 * fin[d * NF]);
                    atomicAdd(fout + d * NF, val);
                }
            }
        }
    }
}

// ---------------------------------------------------------------------------
// Kernel 3: inv_update2 = sum_d f*(f@Wu); add; LayerNorm (in-place on d_out).
// ---------------------------------------------------------------------------
__global__ void __launch_bounds__(NF) finalize_kernel(
    const float* __restrict__ Wu, const float* __restrict__ gamma,
    const float* __restrict__ beta, const float* __restrict__ force_out,
    float* atom_out)
{
    __shared__ float fr[3][NF];
    __shared__ float red[NF];
    const int n = blockIdx.x, t = threadIdx.x;
    #pragma unroll
    for (int d = 0; d < 3; ++d) fr[d][t] = force_out[n * 3 * NF + d * NF + t];
    __syncthreads();

    float inv2 = 0.f;
    #pragma unroll
    for (int d = 0; d < 3; ++d) {
        float g = 0.f;
        #pragma unroll 8
        for (int k = 0; k < NF; ++k) g = fmaf(fr[d][k], Wu[k * NF + t], g);
        inv2 = fmaf(fr[d][t], g, inv2);
    }
    float a = atom_out[n * NF + t] + inv2;

    red[t] = a;
    __syncthreads();
    for (int s = 32; s > 0; s >>= 1) { if (t < s) red[t] += red[t + s]; __syncthreads(); }
    const float mu = red[0] * (1.0f / NF);
    __syncthreads();
    const float dv = a - mu;
    red[t] = dv * dv;
    __syncthreads();
    for (int s = 32; s > 0; s >>= 1) { if (t < s) red[t] += red[t + s]; __syncthreads(); }
    const float var = red[0] * (1.0f / NF);

    atom_out[n * NF + t] = dv * rsqrtf(var + 1e-5f) * gamma[t] + beta[t];
}

// ---------------------------------------------------------------------------
extern "C" void kernel_launch(void* const* d_in, const int* in_sizes, int n_in,
                              void* d_out, int out_size, void* d_ws, size_t ws_size,
                              hipStream_t stream)
{
    const float*     atom      = (const float*)d_in[0];
    const float*     force     = (const float*)d_in[1];
    const float*     dir_edge  = (const float*)d_in[2];
    const float*     dist_edge = (const float*)d_in[3];
    const long long* eidx      = (const long long*)d_in[4];   // int64 per reference
    const float*     W1   = (const float*)d_in[5];
    const float*     b1   = (const float*)d_in[6];
    const float*     W2   = (const float*)d_in[7];
    const float*     b2   = (const float*)d_in[8];
    const float*     We   = (const float*)d_in[9];
    const float*     Wq1a = (const float*)d_in[10];
    const float*     Wq1b = (const float*)d_in[11];
    const float*     Wq2a = (const float*)d_in[12];
    const float*     Wq2b = (const float*)d_in[13];
    const float*     Wu   = (const float*)d_in[14];
    const float*     gamma = (const float*)d_in[15];
    const float*     beta  = (const float*)d_in[16];

    float* atom_out  = (float*)d_out;                              // [N,64]
    float* force_out = atom_out + (size_t)N_NODES * NF;            // [N,3,64]
    float* mn        = (float*)d_ws;                               // [N,64] scratch

    node_mlp_kernel<<<N_NODES, NF, 0, stream>>>(atom, force, W1, b1, W2, b2,
                                                mn, atom_out, force_out);
    // 2500 blocks * (4 waves * 4 tiles * 16 edges) = 640000 edges exactly
    edge_kernel<<<N_EDGES / 256, 128, 0, stream>>>(mn, force, dir_edge, dist_edge, eidx, We,
                                                   Wq1a, Wq1b, Wq2a, Wq2b, atom_out, force_out);
    finalize_kernel<<<N_NODES, NF, 0, stream>>>(Wu, gamma, beta, force_out, atom_out);
}
